// Block_63952063037469
// MI455X (gfx1250) — matrix operand order repaired and checked by reading, not compiled, not run on hardware
//
#include <hip/hip_runtime.h>
#include <math.h>

// ---------------------------------------------------------------------------
// MoE transformer block for MI455X (gfx1250, wave32, WMMA, TDM).
// ln1 -> qkv GEMM (WMMA bf16) -> scalar attention -> proj GEMM (WMMA, fused
// residual) -> ln2 -> top2 gate -> fused MoE MLP (WMMA both layers, 32-token
// tile, 4 experts/block, gate folded into h) -> combine (x1 + partials).
// Weights transposed+converted to bf16 [N][K] once per launch: B fragment =
// one contiguous 32B load/lane. Activation tiles staged to LDS by the TDM.
// ---------------------------------------------------------------------------

#define DIMC   768
#define NTOT   1568       // 8*196 tokens
#define NHEAD  12
#define HDIM   64
#define NEXP   8
#define HIDN   3072
#define SEQ    196
#define BATCH  8
#define MTILES 98         // 1568/16
#define MT32   49         // 1568/32

typedef __attribute__((ext_vector_type(16))) __bf16         v16bf;
typedef __attribute__((ext_vector_type(16))) unsigned short v16u;
typedef __attribute__((ext_vector_type(8)))  unsigned short v8u;
typedef __attribute__((ext_vector_type(8)))  float          v8f;
typedef __attribute__((ext_vector_type(4)))  unsigned       tdm_u4;
typedef __attribute__((ext_vector_type(8)))  int            tdm_i8;
typedef __attribute__((ext_vector_type(4)))  int            tdm_i4;

__device__ __forceinline__ unsigned short f2bf(float f) {
    unsigned u = __float_as_uint(f);
    unsigned r = u + 0x7FFFu + ((u >> 16) & 1u);   // round-to-nearest-even
    return (unsigned short)(r >> 16);
}
__device__ __forceinline__ float bf2f(unsigned short h) {
    return __uint_as_float(((unsigned)h) << 16);
}

// ---- Tensor Data Mover: 1-D contiguous bf16 tile (elems elements) -> LDS ----
#if __has_builtin(__builtin_amdgcn_tensor_load_to_lds)
#define USE_TDM 1
#endif

#ifdef USE_TDM
__device__ __forceinline__ void tdm_load_1d(void* lds_ptr, const void* gptr,
                                            unsigned elems) {
    unsigned long long ga = (unsigned long long)(uintptr_t)gptr;
    unsigned lds_off = (unsigned)(uintptr_t)lds_ptr;  // flat shared: low 32 = LDS offset
    tdm_u4 g0;
    g0[0] = 1u;                                            // count=1 (user D#)
    g0[1] = lds_off;                                       // lds_addr
    g0[2] = (unsigned)(ga & 0xffffffffu);                  // global_addr[31:0]
    g0[3] = ((unsigned)(ga >> 32) & 0x01ffffffu) | 0x80000000u; // addr[56:32] | type=2
    tdm_i8 g1;
    g1[0] = (int)0x00010000u;                  // wg_mask=0, data_size=1 (2 bytes)
    g1[1] = (int)((elems & 0xffffu) << 16);    // tensor_dim0[15:0]
    g1[2] = (int)(((elems >> 16) & 0xffffu) | (1u << 16)); // dim0[31:16] | tensor_dim1=1
    g1[3] = (int)((elems & 0xffffu) << 16);    // tensor_dim1 hi=0 | tile_dim0
    g1[4] = 1;                                 // tile_dim1=1, tile_dim2=0
    g1[5] = (int)elems;                        // tensor_dim0_stride[31:0]
    g1[6] = 0;
    g1[7] = 0;
    tdm_i4 z4 = {0, 0, 0, 0};
#if defined(__clang_major__) && __clang_major__ >= 23
    tdm_i8 z8 = {0, 0, 0, 0, 0, 0, 0, 0};
    __builtin_amdgcn_tensor_load_to_lds(g0, g1, z4, z4, z8, 0);
#else
    __builtin_amdgcn_tensor_load_to_lds(g0, g1, z4, z4, 0);
#endif
}
#endif

// A fragment (16x32 bf16): one contiguous 16-elem run per lane, K = 16h .. 16h+15.
__device__ __forceinline__ v16bf load_a_lds(const unsigned short* base, int ld) {
    int lane = threadIdx.x & 31;
    int m = lane & 15, half = lane >> 4;
    const unsigned short* r0 = base + m * ld + half * 16;
    return __builtin_bit_cast(v16bf, *(const v16u*)r0);
}

// B fragment (32x16) from TRANSPOSED bf16 weights Wt[N][K]: one 32B load/lane.
__device__ __forceinline__ v16bf load_b_gbf16(const unsigned short* Wt, int ldk) {
    int lane = threadIdx.x & 31;
    int n = lane & 15, kh = lane >> 4;
    const unsigned short* p = Wt + (size_t)n * ldk + kh * 16;
    return __builtin_bit_cast(v16bf, *(const v16u*)p);
}

__device__ __forceinline__ v8f wmma_bf16(v16bf a, v16bf b, v8f c) {
    return __builtin_amdgcn_wmma_f32_16x16x32_bf16(
        false, a, false, b, (short)0, c, false, false);
}

// ---------------------------------------------------------------------------
// Weight convert+transpose: W[K][N] f32 -> Wt[N][K] bf16. grid(N/32,K/32,slabs)
// ---------------------------------------------------------------------------
__global__ void __launch_bounds__(256) wconv_kernel(
    const float* __restrict__ W, unsigned short* __restrict__ Wt, int K, int N) {
    __shared__ float t[32][33];
    int tx = threadIdx.x & 31, ty = threadIdx.x >> 5;
    size_t slab = (size_t)blockIdx.z * K * N;
    int n0 = blockIdx.x * 32, k0 = blockIdx.y * 32;
    const float* Wp = W + slab;
    unsigned short* Wtp = Wt + slab;
#pragma unroll
    for (int i = 0; i < 4; ++i)
        t[ty + 8 * i][tx] = Wp[(size_t)(k0 + ty + 8 * i) * N + n0 + tx];
    __syncthreads();
#pragma unroll
    for (int i = 0; i < 4; ++i)
        Wtp[(size_t)(n0 + ty + 8 * i) * K + k0 + tx] = f2bf(t[tx][ty + 8 * i]);
}

// ---------------------------------------------------------------------------
// LayerNorm: one block (256 threads) per token. bf16 out.
// ---------------------------------------------------------------------------
__global__ void ln_kernel(const float* __restrict__ x, const float* __restrict__ g,
                          const float* __restrict__ b, unsigned short* __restrict__ out) {
    int t = blockIdx.x, tid = threadIdx.x;
    const float* row = x + (size_t)t * DIMC;
    __shared__ float red[8];
    __shared__ float mu_s, rs_s;

    float v0 = row[tid], v1 = row[tid + 256], v2 = row[tid + 512];
    float s = v0 + v1 + v2;
#pragma unroll
    for (int m = 16; m > 0; m >>= 1) s += __shfl_xor(s, m, 32);
    if ((tid & 31) == 0) red[tid >> 5] = s;
    __syncthreads();
    if (tid == 0) {
        float tot = 0.f;
        for (int i = 0; i < 8; ++i) tot += red[i];
        mu_s = tot * (1.f / DIMC);
    }
    __syncthreads();
    float mu = mu_s;
    float d0 = v0 - mu, d1 = v1 - mu, d2 = v2 - mu;
    float q = d0 * d0 + d1 * d1 + d2 * d2;
#pragma unroll
    for (int m = 16; m > 0; m >>= 1) q += __shfl_xor(q, m, 32);
    __syncthreads();
    if ((tid & 31) == 0) red[tid >> 5] = q;
    __syncthreads();
    if (tid == 0) {
        float tot = 0.f;
        for (int i = 0; i < 8; ++i) tot += red[i];
        rs_s = rsqrtf(tot * (1.f / DIMC) + 1e-5f);
    }
    __syncthreads();
    float rs = rs_s;
    size_t o = (size_t)t * DIMC;
    out[o + tid]       = f2bf(d0 * rs * g[tid]       + b[tid]);
    out[o + tid + 256] = f2bf(d1 * rs * g[tid + 256] + b[tid + 256]);
    out[o + tid + 512] = f2bf(d2 * rs * g[tid + 512] + b[tid + 512]);
}

// ---------------------------------------------------------------------------
// WMMA GEMM: C[1568,N] = A_bf16[1568,768] @ Wt_bf16[N][768]^T + bias (+res)
// grid (98, N/64), block 128 (4 waves, one 16x16 tile each).
// ---------------------------------------------------------------------------
template <bool RES>
__global__ void __launch_bounds__(128) gemm_kernel(
    const unsigned short* __restrict__ A, const unsigned short* __restrict__ Wt,
    const float* __restrict__ bias, const float* __restrict__ resid,
    float* __restrict__ outF, int N) {
    __shared__ __align__(32) unsigned short As[16 * DIMC];   // 24 KB
    int mt = blockIdx.x, tid = threadIdx.x;

#ifdef USE_TDM
    if (tid < 32) {
        tdm_load_1d(As, A + (size_t)mt * 16 * DIMC, 16 * DIMC);
        __builtin_amdgcn_s_wait_tensorcnt(0);
    }
#else
    {
        const unsigned* Ag = (const unsigned*)(A + (size_t)mt * 16 * DIMC);
        unsigned* Asu = (unsigned*)As;
#pragma unroll
        for (int i = 0; i < 48; ++i) Asu[tid + i * 128] = Ag[tid + i * 128];
    }
#endif
    __syncthreads();

    int wave = tid >> 5, lane = tid & 31;
    int n0 = blockIdx.y * 64 + wave * 16;
    const unsigned short* Wb = Wt + (size_t)n0 * DIMC;
    v8f acc = (v8f)(0.0f);
    for (int kk = 0; kk < DIMC; kk += 32) {
        __builtin_prefetch(Wb + kk + 512, 0, 0);
        v16bf a = load_a_lds(As + kk, DIMC);
        v16bf b = load_b_gbf16(Wb + kk, DIMC);
        acc = wmma_bf16(a, b, acc);
    }
    int col = n0 + (lane & 15);
    int half = lane >> 4;
    float bv = bias[col];
#pragma unroll
    for (int i = 0; i < 8; ++i) {
        int row = mt * 16 + i + half * 8;
        float v = acc[i] + bv;
        if (RES) v += resid[(size_t)row * N + col];
        outF[(size_t)row * N + col] = v;
    }
}

// ---------------------------------------------------------------------------
// Scalar attention: one block (64 threads) per (b,h,query). ~1 GF total.
// ---------------------------------------------------------------------------
__global__ void attn_kernel(const float* __restrict__ qkv,
                            unsigned short* __restrict__ attnO) {
    int qi = blockIdx.x, h = blockIdx.y, b = blockIdx.z;
    __shared__ float qs[HDIM];
    __shared__ float sc[SEQ];
    __shared__ float red[2];
    int tid = threadIdx.x;   // 64 threads

    const float* qp = qkv + (size_t)(b * SEQ + qi) * 2304 + h * 64;
    qs[tid] = qp[tid];
    __syncthreads();

    for (int k = tid; k < SEQ; k += 64) {
        const float* kp = qkv + (size_t)(b * SEQ + k) * 2304 + 768 + h * 64;
        float d = 0.f;
#pragma unroll 8
        for (int j = 0; j < HDIM; ++j) d += qs[j] * kp[j];
        sc[k] = d * 0.125f;   // 1/sqrt(64)
    }
    __syncthreads();
    if (tid == 0) {
        float mx = -1e30f;
        for (int k = 0; k < SEQ; ++k) mx = fmaxf(mx, sc[k]);
        red[0] = mx;
    }
    __syncthreads();
    float mx = red[0];
    for (int k = tid; k < SEQ; k += 64) sc[k] = expf(sc[k] - mx);
    __syncthreads();
    if (tid == 0) {
        float s = 0.f;
        for (int k = 0; k < SEQ; ++k) s += sc[k];
        red[1] = s;
    }
    __syncthreads();
    float inv = 1.f / red[1];

    const float* vp = qkv + (size_t)(b * SEQ) * 2304 + 1536 + h * 64 + tid;
    float o = 0.f;
    for (int k = 0; k < SEQ; ++k) o += sc[k] * vp[(size_t)k * 2304];
    attnO[(size_t)(b * SEQ + qi) * DIMC + h * 64 + tid] = f2bf(o * inv);
}

// ---------------------------------------------------------------------------
// Top-2 gate: one block (256 thr = 8 waves) per token; wave e -> logit e.
// ---------------------------------------------------------------------------
__global__ void gate_kernel(const unsigned short* __restrict__ xn,
                            const float* __restrict__ gw,
                            float* __restrict__ gates) {
    int t = blockIdx.x, tid = threadIdx.x;
    int e = tid >> 5, lane = tid & 31;
    __shared__ float lg[NEXP];
    float s = 0.f;
    for (int j = lane; j < DIMC; j += 32)
        s += bf2f(xn[(size_t)t * DIMC + j]) * gw[(size_t)j * NEXP + e];
#pragma unroll
    for (int m = 16; m > 0; m >>= 1) s += __shfl_xor(s, m, 32);
    if (lane == 0) lg[e] = s;
    __syncthreads();
    if (tid == 0) {
        int i0 = 0;
        for (int i = 1; i < NEXP; ++i) if (lg[i] > lg[i0]) i0 = i;
        int i1 = (i0 == 0) ? 1 : 0;
        for (int i = 0; i < NEXP; ++i) if (i != i0 && lg[i] > lg[i1]) i1 = i;
        float g0 = 1.f / (1.f + expf(lg[i1] - lg[i0]));
        float o[NEXP];
        for (int i = 0; i < NEXP; ++i) o[i] = 0.f;
        o[i0] = g0; o[i1] = 1.f - g0;
        for (int i = 0; i < NEXP; ++i) gates[(size_t)t * NEXP + i] = o[i];
    }
}

// ---------------------------------------------------------------------------
// Fused MoE MLP partial: block (256 thr, 8 waves) = 32-token tile x 4 experts.
// Gate folded into h rows, so fin accumulates across experts directly:
//   part = sum_e (g_e * gelu(xn@w1[e]+b1[e])) @ w2[e] + g_e * b2[e]
// grid (49, 2): blockIdx.y selects experts [4y, 4y+4).
// ---------------------------------------------------------------------------
__global__ void __launch_bounds__(256) moe_kernel(
    const unsigned short* __restrict__ xn,   // [T,768] bf16
    const unsigned short* __restrict__ w1t,  // [E][3072][768] bf16
    const float* __restrict__ b1,            // [E,3072]
    const unsigned short* __restrict__ w2t,  // [E][768][3072] bf16
    const float* __restrict__ b2,            // [E,768]
    const float* __restrict__ gates,         // [T,E]
    float* __restrict__ part) {              // [2][T,768] partial sums
    __shared__ __align__(32) unsigned short Xs[32 * DIMC]; // 48 KB
    __shared__ __align__(32) unsigned short Hs[32 * 128];  //  8 KB
    __shared__ float Gs[32];
    int mt = blockIdx.x, tid = threadIdx.x;
    int wave = tid >> 5, lane = tid & 31, half = lane >> 4;

#ifdef USE_TDM
    if (tid < 32) {
        tdm_load_1d(Xs, xn + (size_t)mt * 32 * DIMC, 32 * DIMC);
        __builtin_amdgcn_s_wait_tensorcnt(0);
    }
#else
    {
        const unsigned* Xg = (const unsigned*)(xn + (size_t)mt * 32 * DIMC);
        unsigned* Xsu = (unsigned*)Xs;
#pragma unroll
        for (int i = 0; i < 48; ++i) Xsu[tid + i * 256] = Xg[tid + i * 256];
    }
#endif
    __syncthreads();

    v8f fin0[6], fin1[6];
#pragma unroll
    for (int s = 0; s < 6; ++s) { fin0[s] = (v8f)(0.0f); fin1[s] = (v8f)(0.0f); }

    int e0 = blockIdx.y * 4;
    for (int ei = 0; ei < 4; ++ei) {
        int e = e0 + ei;
        __syncthreads();
        if (tid < 32) Gs[tid] = gates[(size_t)(mt * 32 + tid) * NEXP + e];
        __syncthreads();

        const unsigned short* W1e = w1t + (size_t)e * DIMC * HIDN;
        const unsigned short* W2e = w2t + (size_t)e * HIDN * DIMC;

        for (int c = 0; c < 24; ++c) {         // hidden chunks of 128
            int h0 = c * 128;
            // Phase A: each wave one 16-col slab of h for BOTH m-tiles (K=768)
            v8f ha0 = (v8f)(0.0f), ha1 = (v8f)(0.0f);
            int nA = h0 + wave * 16;
            const unsigned short* W1b = W1e + (size_t)nA * DIMC;
            for (int kk = 0; kk < DIMC; kk += 32) {
                __builtin_prefetch(W1b + kk + 512, 0, 0);
                v16bf a0 = load_a_lds(Xs + kk, DIMC);
                v16bf a1 = load_a_lds(Xs + 16 * DIMC + kk, DIMC);
                v16bf bb = load_b_gbf16(W1b + kk, DIMC);
                ha0 = wmma_bf16(a0, bb, ha0);
                ha1 = wmma_bf16(a1, bb, ha1);
            }
            __syncthreads();                   // prev chunk's Phase B readers done
            int colA = nA + (lane & 15);
            float b1v = b1[(size_t)e * HIDN + colA];
#pragma unroll
            for (int i = 0; i < 8; ++i) {
                int r0 = i + half * 8;
                float v = ha0[i] + b1v;
                float gl = 0.5f * v * (1.f + erff(v * 0.70710678118f));
                Hs[r0 * 128 + wave * 16 + (lane & 15)] = f2bf(gl * Gs[r0]);
                float w = ha1[i] + b1v;
                float g2 = 0.5f * w * (1.f + erff(w * 0.70710678118f));
                Hs[(16 + r0) * 128 + wave * 16 + (lane & 15)] = f2bf(g2 * Gs[16 + r0]);
            }
            __syncthreads();
            // Phase B: each wave 6 output tiles (96 cols), K = 128, 2 m-tiles
#pragma unroll
            for (int s = 0; s < 6; ++s) {
                int n0 = wave * 96 + s * 16;
                const unsigned short* W2b = W2e + (size_t)n0 * HIDN + h0;
#pragma unroll
                for (int kk = 0; kk < 128; kk += 32) {
                    v16bf a0 = load_a_lds(Hs + kk, 128);
                    v16bf a1 = load_a_lds(Hs + 16 * 128 + kk, 128);
                    v16bf bb = load_b_gbf16(W2b + kk, HIDN);
                    fin0[s] = wmma_bf16(a0, bb, fin0[s]);
                    fin1[s] = wmma_bf16(a1, bb, fin1[s]);
                }
            }
        }
        // per-expert bias term: fin += g[row] * b2[col]
#pragma unroll
        for (int s = 0; s < 6; ++s) {
            int col = wave * 96 + s * 16 + (lane & 15);
            float b2v = b2[(size_t)e * DIMC + col];
#pragma unroll
            for (int i = 0; i < 8; ++i) {
                int r0 = i + half * 8;
                fin0[s][i] += Gs[r0] * b2v;
                fin1[s][i] += Gs[16 + r0] * b2v;
            }
        }
    }
    float* po = part + (size_t)blockIdx.y * NTOT * DIMC;
#pragma unroll
    for (int s = 0; s < 6; ++s) {
        int col = wave * 96 + s * 16 + (lane & 15);
#pragma unroll
        for (int i = 0; i < 8; ++i) {
            int r0 = i + half * 8;
            po[(size_t)(mt * 32 + r0) * DIMC + col]      = fin0[s][i];
            po[(size_t)(mt * 32 + 16 + r0) * DIMC + col] = fin1[s][i];
        }
    }
}

// ---------------------------------------------------------------------------
// Combine: out = x1 + part0 + part1
// ---------------------------------------------------------------------------
__global__ void combine_kernel(const float* __restrict__ x1,
                               const float* __restrict__ part,
                               float* __restrict__ out) {
    size_t i = (size_t)blockIdx.x * 256 + threadIdx.x;
    out[i] = x1[i] + part[i] + part[(size_t)NTOT * DIMC + i];
}

// ---------------------------------------------------------------------------
extern "C" void kernel_launch(void* const* d_in, const int* in_sizes, int n_in,
                              void* d_out, int out_size, void* d_ws, size_t ws_size,
                              hipStream_t stream) {
    (void)in_sizes; (void)n_in; (void)out_size; (void)ws_size;
    const float* x      = (const float*)d_in[0];
    const float* ln1_g  = (const float*)d_in[1];
    const float* ln1_b  = (const float*)d_in[2];
    const float* qkv_w  = (const float*)d_in[3];
    const float* qkv_b  = (const float*)d_in[4];
    const float* proj_w = (const float*)d_in[5];
    const float* proj_b = (const float*)d_in[6];
    const float* ln2_g  = (const float*)d_in[7];
    const float* ln2_b  = (const float*)d_in[8];
    const float* gate_w = (const float*)d_in[9];
    const float* w1     = (const float*)d_in[10];
    const float* b1     = (const float*)d_in[11];
    const float* w2     = (const float*)d_in[12];
    const float* b2     = (const float*)d_in[13];
    float* out = (float*)d_out;

    char* ws = (char*)d_ws;
    size_t off = 0;
    auto carve = [&](size_t bytes) -> void* {
        void* p = ws + off;
        off += (bytes + 255) & ~(size_t)255;
        return p;
    };
    unsigned short* xn1    = (unsigned short*)carve((size_t)NTOT * DIMC * 2);
    float*          qkv    = (float*)         carve((size_t)NTOT * 2304 * 4);
    unsigned short* attnO  = (unsigned short*)carve((size_t)NTOT * DIMC * 2);
    float*          x1v    = (float*)         carve((size_t)NTOT * DIMC * 4);
    unsigned short* xn2    = (unsigned short*)carve((size_t)NTOT * DIMC * 2);
    float*          gts    = (float*)         carve((size_t)NTOT * NEXP * 4);
    float*          partb  = (float*)         carve((size_t)2 * NTOT * DIMC * 4);
    unsigned short* qkvWt  = (unsigned short*)carve((size_t)DIMC * 2304 * 2);
    unsigned short* projWt = (unsigned short*)carve((size_t)DIMC * DIMC * 2);
    unsigned short* w1tp   = (unsigned short*)carve((size_t)NEXP * DIMC * HIDN * 2);
    unsigned short* w2tp   = (unsigned short*)carve((size_t)NEXP * HIDN * DIMC * 2);
    // total scratch ~117 MB

    // Weight convert + transpose (bf16, [N][K])
    wconv_kernel<<<dim3(2304 / 32, DIMC / 32, 1), 256, 0, stream>>>(qkv_w, qkvWt, DIMC, 2304);
    wconv_kernel<<<dim3(DIMC / 32, DIMC / 32, 1), 256, 0, stream>>>(proj_w, projWt, DIMC, DIMC);
    wconv_kernel<<<dim3(HIDN / 32, DIMC / 32, NEXP), 256, 0, stream>>>(w1, w1tp, DIMC, HIDN);
    wconv_kernel<<<dim3(DIMC / 32, HIDN / 32, NEXP), 256, 0, stream>>>(w2, w2tp, HIDN, DIMC);

    ln_kernel<<<NTOT, 256, 0, stream>>>(x, ln1_g, ln1_b, xn1);
    gemm_kernel<false><<<dim3(MTILES, 2304 / 64), 128, 0, stream>>>(
        xn1, qkvWt, qkv_b, nullptr, qkv, 2304);
    attn_kernel<<<dim3(SEQ, NHEAD, BATCH), 64, 0, stream>>>(qkv, attnO);
    gemm_kernel<true><<<dim3(MTILES, DIMC / 64), 128, 0, stream>>>(
        attnO, projWt, proj_b, x, x1v, DIMC);
    ln_kernel<<<NTOT, 256, 0, stream>>>(x1v, ln2_g, ln2_b, xn2);
    gate_kernel<<<NTOT, 256, 0, stream>>>(xn2, gate_w, gts);
    moe_kernel<<<dim3(MT32, 2), 256, 0, stream>>>(xn2, w1tp, b1, w2tp, b2, gts, partb);
    combine_kernel<<<(NTOT * DIMC) / 256, 256, 0, stream>>>(x1v, partb, out);
}
